// SEALAttention_24730421690591
// MI455X (gfx1250) — matrix operand order repaired and checked by reading, not compiled, not run on hardware
//
#include <hip/hip_runtime.h>
#include <hip/hip_bf16.h>

// ---------------------------------------------------------------------------
// SEAL attention for MI455X (gfx1250).
// All matmuls -> v_wmma_f32_16x16x32_f16 (wave32), operands staged in LDS via
// GLOBAL_LOAD_ASYNC_TO_LDS_B128 (ASYNCcnt), double-buffered.
// ---------------------------------------------------------------------------

#define B_C    8
#define L_C    8192
#define D_C    1024
#define SEG_C  256
#define NSEG_C 32
#define M_TOK  (B_C * L_C)          // 65536 tokens
#define SCALE_C 0.125f              // 1/sqrt(64)

#define TILE_M 128
#define TILE_N 128
#define TILE_K 32
#define LDS_ROW 40                  // 32 halves + 8 pad -> 80B row stride:
                                    // b128 fragment reads hit all 64 banks
#define LDS_BUF_BYTES (TILE_M * LDS_ROW * 2)   // 10240B per matrix per buffer

typedef __attribute__((ext_vector_type(16))) _Float16 v16h;
typedef __attribute__((ext_vector_type(8)))  float    v8f;

union H16 { uint4 u[2]; v16h h; };

// ---------------------------------------------------------------------------
// elementwise fp32 -> f16
// ---------------------------------------------------------------------------
__global__ void cvt_f32_f16(const float* __restrict__ src,
                            _Float16* __restrict__ dst, long long n) {
    long long i = ((long long)blockIdx.x * blockDim.x + threadIdx.x) * 4;
    if (i + 3 < n) {
        float4 v = *(const float4*)(src + i);
        dst[i + 0] = (_Float16)v.x;
        dst[i + 1] = (_Float16)v.y;
        dst[i + 2] = (_Float16)v.z;
        dst[i + 3] = (_Float16)v.w;
    } else {
        for (long long j = i; j < n; ++j) dst[j] = (_Float16)src[j];
    }
}

// gather segment summaries: summ[b*NSEG+n, d] = x[b, n*SEG + 0, d]  (as f16)
__global__ void gather_summ(const float* __restrict__ x,
                            _Float16* __restrict__ summ) {
    int row = blockIdx.x;               // b*NSEG + n
    int b = row >> 5, n = row & 31;
    const float* src = x + ((size_t)b * L_C + (size_t)n * SEG_C) * D_C;
    for (int d = threadIdx.x; d < D_C; d += blockDim.x)
        summ[(size_t)row * D_C + d] = (_Float16)src[d];
}

// ---------------------------------------------------------------------------
// Batched WMMA GEMM:  C[bat] = (A[bat] * Bm[bat]^T + bias[bat]) * scale
//   A   : (M,K) f16 row-major            (per-batch stride sA elements)
//   Bm  : (N,K) f16 row-major (k-contig) (per-batch stride sB)
//   bias: f32 length-N, per-batch stride sBias (may be null)
//   OUTF32: f32 output, else f16; SEGT>0: chunk-transposed f16 store
// 256 threads = 8 wave32s; block tile 128x128 staged in LDS (double-buffered
// async copies); wave tile 32x64 = 2x4 WMMA fragments.
// ---------------------------------------------------------------------------
template <int SEGT, bool OUTF32>
__global__ __launch_bounds__(256)
void wmma_gemm(const _Float16* __restrict__ A,  long long sA,
               const _Float16* __restrict__ Bm, long long sB,
               const float*    __restrict__ bias, long long sBias,
               void* __restrict__ Cout, long long sC,
               int M, int N, int K, float scale)
{
    __shared__ __align__(16) _Float16 As[2][TILE_M * LDS_ROW];
    __shared__ __align__(16) _Float16 Bs[2][TILE_N * LDS_ROW];

    const int batch = blockIdx.z;
    A  += (size_t)batch * sA;
    Bm += (size_t)batch * sB;

    const int tid  = threadIdx.x;
    const int w    = tid >> 5;
    const int lane = tid & 31;
    const int lrow = lane & 15;          // row/col within fragment
    const int lhi  = lane >> 4;          // K-half selector
    const int mBase = blockIdx.x * TILE_M;
    const int nBase = blockIdx.y * TILE_N;
    const int wm = (w & 3) * 32;         // wave origin inside block tile
    const int wn = (w >> 2) * 64;

    // ---- per-thread async-copy slots (hoisted out of the k loop) ----
    // 16B chunk c = r*4 + part ; this thread owns chunks tid and tid+256.
    const int part = tid & 3;
    const int r0   = tid >> 2;           // rows r0 and r0+64 of the tile
    int grA0 = mBase + r0;      grA0 = grA0 < M ? grA0 : M - 1;  // clamp keeps
    int grA1 = mBase + r0 + 64; grA1 = grA1 < M ? grA1 : M - 1;  // EXEC all-1s
    int gcB0 = nBase + r0;      gcB0 = gcB0 < N ? gcB0 : N - 1;
    int gcB1 = nBase + r0 + 64; gcB1 = gcB1 < N ? gcB1 : N - 1;
    const _Float16* srcA0 = A  + (size_t)grA0 * K + part * 8;
    const _Float16* srcA1 = A  + (size_t)grA1 * K + part * 8;
    const _Float16* srcB0 = Bm + (size_t)gcB0 * K + part * 8;
    const _Float16* srcB1 = Bm + (size_t)gcB1 * K + part * 8;
    const unsigned ldsA0 = (unsigned)(size_t)&As[0][r0 * LDS_ROW + part * 8];
    const unsigned ldsA1 = ldsA0 + 64u * LDS_ROW * 2u;
    const unsigned ldsB0 = (unsigned)(size_t)&Bs[0][r0 * LDS_ROW + part * 8];
    const unsigned ldsB1 = ldsB0 + 64u * LDS_ROW * 2u;

    auto issue_tile = [&](int buf, int k0) {
        const unsigned bo = buf ? (unsigned)LDS_BUF_BYTES : 0u;
        asm volatile("global_load_async_to_lds_b128 %0, %1, off"
                     :: "v"(ldsA0 + bo), "v"(srcA0 + k0) : "memory");
        asm volatile("global_load_async_to_lds_b128 %0, %1, off"
                     :: "v"(ldsA1 + bo), "v"(srcA1 + k0) : "memory");
        asm volatile("global_load_async_to_lds_b128 %0, %1, off"
                     :: "v"(ldsB0 + bo), "v"(srcB0 + k0) : "memory");
        asm volatile("global_load_async_to_lds_b128 %0, %1, off"
                     :: "v"(ldsB1 + bo), "v"(srcB1 + k0) : "memory");
    };

    issue_tile(0, 0);

    v8f acc[2][4];
    #pragma unroll
    for (int mi = 0; mi < 2; ++mi)
        #pragma unroll
        for (int ni = 0; ni < 4; ++ni) {
            float bv = 0.0f;
            if (bias) {
                int col = nBase + wn + ni * 16 + lrow;
                col = col < N ? col : N - 1;
                bv = bias[(size_t)batch * sBias + col];
            }
            #pragma unroll
            for (int r = 0; r < 8; ++r) acc[mi][ni][r] = bv;
        }

    asm volatile("s_wait_asynccnt 0" ::: "memory");
    __syncthreads();

    const int nk = K / TILE_K;
    for (int kt = 0; kt < nk; ++kt) {
        const int cur = kt & 1;
        if (kt + 1 < nk) issue_tile(cur ^ 1, (kt + 1) * TILE_K);

        // fragment reads from LDS (ds_load_b128, bank-conflict-free via pad)
        H16 af[2], bf[4];
        #pragma unroll
        for (int mi = 0; mi < 2; ++mi) {
            const _Float16* p = &As[cur][(wm + mi * 16 + lrow) * LDS_ROW + lhi * 8];
            af[mi].u[0] = *(const uint4*)(p);        // K halves lhi*8 .. +7
            af[mi].u[1] = *(const uint4*)(p + 16);   // K halves 16+lhi*8 .. +7
        }
        #pragma unroll
        for (int ni = 0; ni < 4; ++ni) {
            const _Float16* q = &Bs[cur][(wn + ni * 16 + lrow) * LDS_ROW + lhi * 8];
            bf[ni].u[0] = *(const uint4*)(q);        // K halves lhi*8 .. +7
            bf[ni].u[1] = *(const uint4*)(q + 16);   // K halves 16+lhi*8 .. +7
        }
        #pragma unroll
        for (int mi = 0; mi < 2; ++mi)
            #pragma unroll
            for (int ni = 0; ni < 4; ++ni)
                acc[mi][ni] = __builtin_amdgcn_wmma_f32_16x16x32_f16(
                    false, af[mi].h, false, bf[ni].h,
                    (short)0, acc[mi][ni], false, false);

        if (kt + 1 < nk)
            asm volatile("s_wait_asynccnt 0" ::: "memory");
        __syncthreads();   // all waves done reading `cur` + next buffer ready
    }

    // D layout: VGPR r -> row (+mi*16+lhi*8+r), col (+ni*16+lrow)
    #pragma unroll
    for (int mi = 0; mi < 2; ++mi)
        #pragma unroll
        for (int ni = 0; ni < 4; ++ni) {
            int col = nBase + wn + ni * 16 + lrow;
            if (col >= N) continue;
            #pragma unroll
            for (int r = 0; r < 8; ++r) {
                int row = mBase + wm + mi * 16 + lhi * 8 + r;
                if (row >= M) continue;
                float v = acc[mi][ni][r] * scale;
                if constexpr (OUTF32) {
                    ((float*)Cout)[(size_t)batch * sC + (size_t)row * N + col] = v;
                } else if constexpr (SEGT > 0) {
                    // chunk-transposed: (row/SEGT, col, row%SEGT); SEGT pow2
                    size_t idx = (size_t)(row / SEGT) * ((size_t)N * SEGT)
                               + (size_t)col * SEGT + (size_t)(row & (SEGT - 1));
                    ((_Float16*)Cout)[idx] = (_Float16)v;
                } else {
                    ((_Float16*)Cout)[(size_t)batch * sC + (size_t)row * N + col]
                        = (_Float16)v;
                }
            }
        }
}

// ---------------------------------------------------------------------------
// row softmax: one 256-thread block per row (len = 256 or 32), f16 output
// ---------------------------------------------------------------------------
__global__ __launch_bounds__(256)
void softmax_rows(const float* __restrict__ S, _Float16* __restrict__ P, int len) {
    const int row = blockIdx.x;
    const int t   = threadIdx.x;
    const float* s = S + (size_t)row * len;
    __shared__ float red[256];

    float v = (t < len) ? s[t] : -3.4e38f;
    red[t] = v; __syncthreads();
    #pragma unroll
    for (int off = 128; off > 0; off >>= 1) {
        if (t < off) red[t] = fmaxf(red[t], red[t + off]);
        __syncthreads();
    }
    float m = red[0]; __syncthreads();

    float e = (t < len) ? __expf(v - m) : 0.0f;
    red[t] = e; __syncthreads();
    #pragma unroll
    for (int off = 128; off > 0; off >>= 1) {
        if (t < off) red[t] += red[t + off];
        __syncthreads();
    }
    float inv = 1.0f / red[0];
    if (t < len) P[(size_t)row * len + t] = (_Float16)(e * inv);
}

// ---------------------------------------------------------------------------
extern "C" void kernel_launch(void* const* d_in, const int* in_sizes, int n_in,
                              void* d_out, int out_size, void* d_ws, size_t ws_size,
                              hipStream_t stream) {
    const float* x  = (const float*)d_in[0];
    const float* wq = (const float*)d_in[1];
    const float* bq = (const float*)d_in[2];
    const float* wk = (const float*)d_in[3];
    const float* bk = (const float*)d_in[4];
    const float* wv = (const float*)d_in[5];
    const float* bv = (const float*)d_in[6];
    const float* wc = (const float*)d_in[7];
    const float* bc = (const float*)d_in[8];
    float* out = (float*)d_out;

    // ---- workspace carve-up (256B aligned) ----
    char* ws = (char*)d_ws;
    size_t cur = 0;
    auto take = [&](size_t bytes) -> void* {
        void* p = ws + cur;
        cur += (bytes + 255) & ~(size_t)255;
        return p;
    };
    const long long XN = (long long)M_TOK * D_C;              // 67,108,864
    const long long WN = (long long)D_C * D_C;
    _Float16* xh    = (_Float16*)take(XN * 2);   // reused as fused_h later
    _Float16* wqh   = (_Float16*)take(WN * 2);
    _Float16* wkh   = (_Float16*)take(WN * 2);
    _Float16* wvh   = (_Float16*)take(WN * 2);
    _Float16* wch   = (_Float16*)take(WN * 2);
    _Float16* Qh    = (_Float16*)take(XN * 2);
    _Float16* Kh    = (_Float16*)take(XN * 2);
    _Float16* Vt    = (_Float16*)take(XN * 2);   // (b,n) chunks: (e, t) e-major
    float*    Sc    = (float*)   take((size_t)256 * SEG_C * SEG_C * 4);
    _Float16* Ph    = (_Float16*)take((size_t)256 * SEG_C * SEG_C * 2);
    _Float16* summh = (_Float16*)take((size_t)256 * D_C * 2);
    _Float16* Qgh   = (_Float16*)take((size_t)256 * D_C * 2);
    _Float16* Kgh   = (_Float16*)take((size_t)256 * D_C * 2);
    _Float16* Vgt   = (_Float16*)take((size_t)256 * D_C * 2);
    float*    Sg    = (float*)   take((size_t)B_C * NSEG_C * NSEG_C * 4);
    _Float16* Pg    = (_Float16*)take((size_t)B_C * NSEG_C * NSEG_C * 2);
    float*    gout  = (float*)   take((size_t)256 * D_C * 4);
    _Float16* fusedh = xh;   // x f16 dead once Q/K/V are built

    const dim3 blk(256);

    // 1) convert x + weights to f16; gather summaries
    cvt_f32_f16<<<dim3((XN / 4 + 255) / 256), blk, 0, stream>>>(x,  xh,  XN);
    cvt_f32_f16<<<dim3((WN / 4 + 255) / 256), blk, 0, stream>>>(wq, wqh, WN);
    cvt_f32_f16<<<dim3((WN / 4 + 255) / 256), blk, 0, stream>>>(wk, wkh, WN);
    cvt_f32_f16<<<dim3((WN / 4 + 255) / 256), blk, 0, stream>>>(wv, wvh, WN);
    cvt_f32_f16<<<dim3((WN / 4 + 255) / 256), blk, 0, stream>>>(wc, wch, WN);
    gather_summ<<<dim3(256), blk, 0, stream>>>(x, summh);

    // 2) Q/K/V projections (M=65536, N=K=1024); V stored chunk-transposed
    const dim3 gBig(M_TOK / TILE_M, D_C / TILE_N, 1);
    wmma_gemm<0, false><<<gBig, blk, 0, stream>>>(
        xh, 0, wqh, 0, bq, 0, Qh, 0, M_TOK, D_C, D_C, 1.0f);
    wmma_gemm<0, false><<<gBig, blk, 0, stream>>>(
        xh, 0, wkh, 0, bk, 0, Kh, 0, M_TOK, D_C, D_C, 1.0f);
    wmma_gemm<SEG_C, false><<<gBig, blk, 0, stream>>>(
        xh, 0, wvh, 0, bv, 0, Vt, 0, M_TOK, D_C, D_C, 1.0f);

    // 3) global-path projections (M=256)
    const dim3 gSm(2, D_C / TILE_N, 1);
    wmma_gemm<0, false><<<gSm, blk, 0, stream>>>(
        summh, 0, wqh, 0, bq, 0, Qgh, 0, 256, D_C, D_C, 1.0f);
    wmma_gemm<0, false><<<gSm, blk, 0, stream>>>(
        summh, 0, wkh, 0, bk, 0, Kgh, 0, 256, D_C, D_C, 1.0f);
    wmma_gemm<NSEG_C, false><<<gSm, blk, 0, stream>>>(
        summh, 0, wvh, 0, bv, 0, Vgt, 0, 256, D_C, D_C, 1.0f);

    // 4) local scores: batch=256 of (256x256, K=1024), *SCALE, f32 out
    wmma_gemm<0, true><<<dim3(2, 2, 256), blk, 0, stream>>>(
        Qh, (long long)SEG_C * D_C, Kh, (long long)SEG_C * D_C,
        nullptr, 0, Sc, (long long)SEG_C * SEG_C,
        SEG_C, SEG_C, D_C, SCALE_C);

    // 5) global scores: batch=8 of (32x32, K=1024)
    wmma_gemm<0, true><<<dim3(1, 1, B_C), blk, 0, stream>>>(
        Qgh, (long long)NSEG_C * D_C, Kgh, (long long)NSEG_C * D_C,
        nullptr, 0, Sg, (long long)NSEG_C * NSEG_C,
        NSEG_C, NSEG_C, D_C, SCALE_C);

    // 6) softmaxes -> f16 probabilities
    softmax_rows<<<dim3(256 * SEG_C), blk, 0, stream>>>(Sc, Ph, SEG_C);
    softmax_rows<<<dim3(B_C * NSEG_C), blk, 0, stream>>>(Sg, Pg, NSEG_C);

    // 7) global attn*V -> gout (f32), batch=8, M=32, N=1024, K=32
    wmma_gemm<0, true><<<dim3(1, D_C / TILE_N, B_C), blk, 0, stream>>>(
        Pg, (long long)NSEG_C * NSEG_C, Vgt, (long long)D_C * NSEG_C,
        nullptr, 0, gout, (long long)NSEG_C * D_C,
        NSEG_C, D_C, NSEG_C, 1.0f);

    // 8) local attn*V + gout (as per-batch bias) -> fused f16
    wmma_gemm<0, false><<<dim3(2, D_C / TILE_N, 256), blk, 0, stream>>>(
        Ph, (long long)SEG_C * SEG_C, Vt, (long long)D_C * SEG_C,
        gout, (long long)D_C, fusedh, (long long)SEG_C * D_C,
        SEG_C, D_C, SEG_C, 1.0f);

    // 9) output projection: fused @ wc^T + bc -> f32 d_out
    wmma_gemm<0, true><<<gBig, blk, 0, stream>>>(
        fusedh, 0, wch, 0, bc, 0, out, 0, M_TOK, D_C, D_C, 1.0f);
}
